// OEHeadBlock_51737176047940
// MI455X (gfx1250) — matrix operand order repaired and checked by reading, not compiled, not run on hardware
//
#include <hip/hip_runtime.h>
#include <hip/hip_bf16.h>
#include <math.h>

typedef __bf16 bf16;
typedef __attribute__((ext_vector_type(16))) __bf16 v16bf;
typedef __attribute__((ext_vector_type(8)))  __bf16 v8bf;
typedef __attribute__((ext_vector_type(8)))  float  v8f;

#define B_     16
#define C_     256
#define H_     56
#define W_     56
#define NTOK   3136      // 56*56
#define MROWS  50176     // B_*NTOK
#define NH_    8
#define HD_    32
#define WSQ    49
#define NWIN   1024      // B_ * 8*8 windows
#define HID_   1024

__device__ __forceinline__ float gelu_f(float x) {
    return 0.5f * x * (1.0f + erff(x * 0.70710678118654752440f));
}

// ---------------------------------------------------------------------------
// f32 -> bf16 convert (weights)
// ---------------------------------------------------------------------------
__global__ void cvt_bf16_kernel(const float* __restrict__ in, bf16* __restrict__ out, int n) {
    int i = blockIdx.x * 256 + threadIdx.x;
    if (i < n) out[i] = (bf16)in[i];
}

__global__ void zero_f32_kernel(float* __restrict__ p, int n) {
    int i = blockIdx.x * 256 + threadIdx.x;
    if (i < n) p[i] = 0.0f;
}

// ---------------------------------------------------------------------------
// LN1 + window partition + LePE positional convs (per 7x7 window)
// out: tbuf[(win*49+t)*256 + c] (bf16, window-ordered), r1 = raw xl residual
// ---------------------------------------------------------------------------
__global__ void __launch_bounds__(256) ln1_pos_kernel(
    const float* __restrict__ x, const float* __restrict__ lw, const float* __restrict__ lb,
    const float* __restrict__ pw1, const float* __restrict__ pw2,
    float* __restrict__ r1, bf16* __restrict__ tbuf)
{
    const int win = blockIdx.x;
    const int c   = threadIdx.x;
    const int b   = win >> 6;
    const int wi  = win & 63;
    const int wy  = wi >> 3, wx = wi & 7;

    __shared__ float sv[WSQ * C_];
    __shared__ float smean[WSQ], srstd[WSQ];

    float xv[WSQ];
    const float* xc = x + ((size_t)(b * C_ + c)) * (H_ * W_);
    #pragma unroll
    for (int t = 0; t < WSQ; t++) {
        int iy = t / 7, ix = t % 7;
        int h = wy * 7 + iy, w = wx * 7 + ix;
        float val = xc[h * W_ + w];
        xv[t] = val;
        sv[t * C_ + c] = val;
        r1[((size_t)(b * NTOK + h * W_ + w)) * C_ + c] = val;
    }
    __syncthreads();

    const int wavei = c >> 5, lane = c & 31;
    for (int t = wavei; t < WSQ; t += 8) {
        float s = 0.0f, s2 = 0.0f;
        #pragma unroll
        for (int j = 0; j < 8; j++) {
            float u = sv[t * C_ + lane + j * 32];
            s += u; s2 += u * u;
        }
        #pragma unroll
        for (int off = 16; off > 0; off >>= 1) {
            s  += __shfl_xor(s,  off, 32);
            s2 += __shfl_xor(s2, off, 32);
        }
        if (lane == 0) {
            float m  = s * (1.0f / C_);
            float va = s2 * (1.0f / C_) - m * m;
            smean[t] = m;
            srstd[t] = rsqrtf(va + 1e-6f);
        }
    }
    __syncthreads();

    const float lwc = lw[c], lbc = lb[c];
    float vv[WSQ];
    #pragma unroll
    for (int t = 0; t < WSQ; t++)
        vv[t] = (xv[t] - smean[t]) * srstd[t] * lwc + lbc;

    float k1[9], k2[9];
    #pragma unroll
    for (int j = 0; j < 9; j++) { k1[j] = pw1[c * 9 + j]; k2[j] = pw2[c * 9 + j]; }

    // dwconv1 (SAME, zero pad at window edge) + exact GELU
    float gv[WSQ];
    #pragma unroll
    for (int iy = 0; iy < 7; iy++) {
        #pragma unroll
        for (int ix = 0; ix < 7; ix++) {
            float a = 0.0f;
            #pragma unroll
            for (int dy = 0; dy < 3; dy++) {
                #pragma unroll
                for (int dx = 0; dx < 3; dx++) {
                    int yy = iy + dy - 1, xx = ix + dx - 1;
                    if (yy >= 0 && yy < 7 && xx >= 0 && xx < 7)
                        a += k1[dy * 3 + dx] * vv[yy * 7 + xx];
                }
            }
            gv[iy * 7 + ix] = gelu_f(a);
        }
    }
    // out = vv + dwconv2(gv)
    bf16* tr = tbuf + ((size_t)win * WSQ) * C_ + c;
    #pragma unroll
    for (int iy = 0; iy < 7; iy++) {
        #pragma unroll
        for (int ix = 0; ix < 7; ix++) {
            float a = vv[iy * 7 + ix];
            #pragma unroll
            for (int dy = 0; dy < 3; dy++) {
                #pragma unroll
                for (int dx = 0; dx < 3; dx++) {
                    int yy = iy + dy - 1, xx = ix + dx - 1;
                    if (yy >= 0 && yy < 7 && xx >= 0 && xx < 7)
                        a += k2[dy * 3 + dx] * gv[yy * 7 + xx];
                }
            }
            tr[(size_t)(iy * 7 + ix) * C_] = (bf16)a;
        }
    }
}

// ---------------------------------------------------------------------------
// Register-blocked bf16 WMMA GEMM: out[m][n] = sum_k A[m][k]*W[n][k] + bias[n]
// A: MxK bf16 row-major, W: NxK bf16 row-major (B = W^T).
// Each wave computes a 32x64 output tile (2 M-frags x 4 N-frags = 8 wmma per
// 32-wide K step): A frags reused 4x, B frags 2x -> 24 B/lane operand traffic
// per v_wmma_f32_16x16x32_bf16 (vs 64 B unblocked). Block = 4 waves = 64x128.
// global_prefetch_b8 issued for the next K panel of A (speculative).
// ---------------------------------------------------------------------------
__global__ void __launch_bounds__(128) gemm_wmma_bf16(
    const bf16* __restrict__ A, const bf16* __restrict__ Wt,
    const float* __restrict__ bias, const float* __restrict__ resid,
    float* __restrict__ outF, bf16* __restrict__ outH,
    int M, int N, int K)
{
    const int lane = threadIdx.x;                 // 0..31
    const int wy   = threadIdx.y >> 1;            // 0..1 (M direction)
    const int wx   = threadIdx.y & 1;             // 0..1 (N direction)
    const int m0   = blockIdx.x * 64  + wy * 32;  // wave's 32-row strip
    const int n0   = blockIdx.y * 128 + wx * 64;  // wave's 64-col strip
    const int koffA = (lane < 16) ? 0 : 16;       // contiguous k = 16h + s (matches B)
    const int koffB = (lane < 16) ? 0 : 16;       // contiguous k = 16h + s (matches A)

    const bf16* Ap0 = A + (size_t)(m0 + (lane & 15)) * K;
    const bf16* Ap1 = Ap0 + (size_t)16 * K;
    const bf16* Wp0 = Wt + (size_t)(n0 +  0 + (lane & 15)) * K;
    const bf16* Wp1 = Wt + (size_t)(n0 + 16 + (lane & 15)) * K;
    const bf16* Wp2 = Wt + (size_t)(n0 + 32 + (lane & 15)) * K;
    const bf16* Wp3 = Wt + (size_t)(n0 + 48 + (lane & 15)) * K;

    v8f acc[2][4];
    #pragma unroll
    for (int mt = 0; mt < 2; mt++)
        #pragma unroll
        for (int nt = 0; nt < 4; nt++)
            #pragma unroll
            for (int i = 0; i < 8; i++) acc[mt][nt][i] = 0.0f;

    for (int k0 = 0; k0 < K; k0 += 32) {
        // speculative prefetch of the next A K-panel (gfx1250 global_prefetch_b8)
        __builtin_prefetch(Ap0 + k0 + 128, 0, 1);
        __builtin_prefetch(Ap1 + k0 + 128, 0, 1);

        v16bf a[2];
        {
            v8bf lo0 = *(const v8bf*)(Ap0 + k0 + koffA);
            v8bf hi0 = *(const v8bf*)(Ap0 + k0 + 8 + koffA);
            v8bf lo1 = *(const v8bf*)(Ap1 + k0 + koffA);
            v8bf hi1 = *(const v8bf*)(Ap1 + k0 + 8 + koffA);
            #pragma unroll
            for (int i = 0; i < 8; i++) {
                a[0][i] = lo0[i]; a[0][i + 8] = hi0[i];
                a[1][i] = lo1[i]; a[1][i + 8] = hi1[i];
            }
        }
        v16bf bm[4];
        bm[0] = *(const v16bf*)(Wp0 + k0 + koffB);
        bm[1] = *(const v16bf*)(Wp1 + k0 + koffB);
        bm[2] = *(const v16bf*)(Wp2 + k0 + koffB);
        bm[3] = *(const v16bf*)(Wp3 + k0 + koffB);

        #pragma unroll
        for (int mt = 0; mt < 2; mt++)
            #pragma unroll
            for (int nt = 0; nt < 4; nt++)
                acc[mt][nt] = __builtin_amdgcn_wmma_f32_16x16x32_bf16(
                    false, a[mt], false, bm[nt], (short)0, acc[mt][nt], false, false);
    }

    const int roff = (lane < 16) ? 0 : 8;
    #pragma unroll
    for (int nt = 0; nt < 4; nt++) {
        const int col = n0 + nt * 16 + (lane & 15);
        const float bv = bias ? bias[col] : 0.0f;
        #pragma unroll
        for (int mt = 0; mt < 2; mt++) {
            const int rbase = m0 + mt * 16 + roff;
            #pragma unroll
            for (int r = 0; r < 8; r++) {
                const size_t idx = (size_t)(rbase + r) * N + col;
                float v = acc[mt][nt][r] + bv;
                if (outH) outH[idx] = (bf16)v;
                else      outF[idx] = v + (resid ? resid[idx] : 0.0f);
            }
        }
    }
}

// ---------------------------------------------------------------------------
// Windowed attention: per (window, head). 49x49 scores + rel-pos bias +
// softmax + AV. Output written de-windowed (natural (b, h*56+w) row order)
// so the proj GEMM can fuse the residual.
// ---------------------------------------------------------------------------
__global__ void __launch_bounds__(64) attn_kernel(
    const bf16* __restrict__ qkv, const float* __restrict__ rpb,
    bf16* __restrict__ obuf)
{
    const int win  = blockIdx.x;
    const int head = blockIdx.y;
    const int tid  = threadIdx.x;

    __shared__ float q[WSQ][HD_], k[WSQ][HD_], v[WSQ][HD_];
    for (int idx = tid; idx < WSQ * HD_; idx += 64) {
        int t = idx / HD_, d = idx % HD_;
        size_t base = ((size_t)(win * WSQ + t)) * (3 * C_) + head * HD_ + d;
        q[t][d] = (float)qkv[base];
        k[t][d] = (float)qkv[base + C_];
        v[t][d] = (float)qkv[base + 2 * C_];
    }
    __syncthreads();

    const int i = tid;
    if (i < WSQ) {
        const float scale = 0.17677669529663688f;   // 1/sqrt(32)
        const int ci = 13 * (i / 7) + (i % 7);
        float s[WSQ];
        float mx = -3.0e38f;
        #pragma unroll
        for (int j = 0; j < WSQ; j++) {
            float dot = 0.0f;
            #pragma unroll
            for (int d = 0; d < HD_; d++) dot += q[i][d] * k[j][d];
            int jr = 48 - j;
            int cj = 13 * (jr / 7) + (jr % 7);
            s[j] = dot * scale + rpb[(ci + cj) * NH_ + head];
            mx = fmaxf(mx, s[j]);
        }
        float sum = 0.0f;
        #pragma unroll
        for (int j = 0; j < WSQ; j++) { s[j] = expf(s[j] - mx); sum += s[j]; }
        const float inv = 1.0f / sum;

        // de-windowed output row
        const int b = win >> 6, wi = win & 63, wy = wi >> 3, wx = wi & 7;
        const int iy = i / 7, ix = i % 7;
        const int m = b * NTOK + (wy * 7 + iy) * W_ + (wx * 7 + ix);
        bf16* orow = obuf + (size_t)m * C_ + head * HD_;
        #pragma unroll
        for (int d = 0; d < HD_; d++) {
            float a = 0.0f;
            #pragma unroll
            for (int j = 0; j < WSQ; j++) a += s[j] * v[j][d];
            orow[d] = (bf16)(a * inv);
        }
    }
}

// ---------------------------------------------------------------------------
// LN2: one wave per token row (256 channels), f32 in -> bf16 out
// ---------------------------------------------------------------------------
__global__ void __launch_bounds__(256) ln2_kernel(
    const float* __restrict__ xin, const float* __restrict__ lw,
    const float* __restrict__ lb, bf16* __restrict__ out)
{
    const int wavei = threadIdx.x >> 5, lane = threadIdx.x & 31;
    const int m = blockIdx.x * 8 + wavei;
    const float* row = xin + (size_t)m * C_;
    float u[8];
    float s = 0.0f, s2 = 0.0f;
    #pragma unroll
    for (int j = 0; j < 8; j++) {
        u[j] = row[lane + j * 32];
        s += u[j]; s2 += u[j] * u[j];
    }
    #pragma unroll
    for (int off = 16; off > 0; off >>= 1) {
        s  += __shfl_xor(s,  off, 32);
        s2 += __shfl_xor(s2, off, 32);
    }
    const float mean = s * (1.0f / C_);
    const float rstd = rsqrtf(s2 * (1.0f / C_) - mean * mean + 1e-6f);
    bf16* orow = out + (size_t)m * C_;
    #pragma unroll
    for (int j = 0; j < 8; j++) {
        int c = lane + j * 32;
        orow[c] = (bf16)((u[j] - mean) * rstd * lw[c] + lb[c]);
    }
}

// ---------------------------------------------------------------------------
// BatchNorm statistics: per-channel sum & sumsq over M rows (atomic finish)
// ---------------------------------------------------------------------------
__global__ void __launch_bounds__(256) bn_stats_bf16_kernel(
    const bf16* __restrict__ X, float* __restrict__ stats, int Cch, int rows)
{
    __shared__ float ss[2 * HID_];
    const int tid = threadIdx.x;
    for (int c = tid; c < Cch; c += 256) { ss[c] = 0.0f; ss[Cch + c] = 0.0f; }
    __syncthreads();
    const size_t r0 = (size_t)blockIdx.x * rows;
    for (int r = 0; r < rows; r++) {
        const bf16* row = X + (r0 + r) * Cch;
        for (int c = tid; c < Cch; c += 256) {
            float v = (float)row[c];
            ss[c] += v; ss[Cch + c] += v * v;
        }
    }
    __syncthreads();
    for (int c = tid; c < Cch; c += 256) {
        atomicAdd(&stats[c], ss[c]);
        atomicAdd(&stats[Cch + c], ss[Cch + c]);
    }
}

__global__ void __launch_bounds__(256) bn_stats_f32_kernel(
    const float* __restrict__ X, float* __restrict__ stats, int Cch, int rows)
{
    __shared__ float ss[2 * HID_];
    const int tid = threadIdx.x;
    for (int c = tid; c < Cch; c += 256) { ss[c] = 0.0f; ss[Cch + c] = 0.0f; }
    __syncthreads();
    const size_t r0 = (size_t)blockIdx.x * rows;
    for (int r = 0; r < rows; r++) {
        const float* row = X + (r0 + r) * Cch;
        for (int c = tid; c < Cch; c += 256) {
            float v = row[c];
            ss[c] += v; ss[Cch + c] += v * v;
        }
    }
    __syncthreads();
    for (int c = tid; c < Cch; c += 256) {
        atomicAdd(&stats[c], ss[c]);
        atomicAdd(&stats[Cch + c], ss[Cch + c]);
    }
}

__global__ void bn_finalize_kernel(
    const float* __restrict__ stats, const float* __restrict__ g,
    const float* __restrict__ b, float* __restrict__ scsh, int Cch, float inv_n)
{
    int c = blockIdx.x * 256 + threadIdx.x;
    if (c < Cch) {
        float mean = stats[c] * inv_n;
        float var  = stats[Cch + c] * inv_n - mean * mean;
        float sc   = g[c] * rsqrtf(var + 1e-5f);
        scsh[c]        = sc;
        scsh[Cch + c]  = b[c] - mean * sc;
    }
}

__global__ void __launch_bounds__(256) bn_gelu_bf16_kernel(
    const bf16* __restrict__ X, const float* __restrict__ scsh,
    bf16* __restrict__ Y, int Cch)
{
    size_t idx = (size_t)blockIdx.x * 256 + threadIdx.x;
    int c = (int)(idx & (size_t)(Cch - 1));     // Cch is a power of two
    float v = (float)X[idx] * scsh[c] + scsh[Cch + c];
    Y[idx] = (bf16)gelu_f(v);
}

// ---------------------------------------------------------------------------
// Depthwise 3x3 conv (SAME) over full 56x56, NHWC bf16, +bias
// ---------------------------------------------------------------------------
__global__ void __launch_bounds__(256) dwconv_kernel(
    const bf16* __restrict__ X, const float* __restrict__ wgt,
    const float* __restrict__ bias, bf16* __restrict__ Y)
{
    const int p = blockIdx.x;                 // b*3136 + y*56 + x
    const int b = p / NTOK;
    const int s = p % NTOK;
    const int y = s / W_, x = s % W_;
    const int tid = threadIdx.x;
    #pragma unroll
    for (int kk = 0; kk < 4; kk++) {
        const int c = tid + kk * 256;
        float acc = bias[c];
        #pragma unroll
        for (int dy = 0; dy < 3; dy++) {
            #pragma unroll
            for (int dx = 0; dx < 3; dx++) {
                int yy = y + dy - 1, xx = x + dx - 1;
                if (yy >= 0 && yy < H_ && xx >= 0 && xx < W_) {
                    size_t idx = ((size_t)(b * NTOK + yy * W_ + xx)) * HID_ + c;
                    acc += wgt[c * 9 + dy * 3 + dx] * (float)X[idx];
                }
            }
        }
        Y[(size_t)p * HID_ + c] = (bf16)acc;
    }
}

// ---------------------------------------------------------------------------
// Final: d_out[b][c][h][w] = r2 + gelu(bn3(g3)), NLC -> NCHW
// ---------------------------------------------------------------------------
__global__ void __launch_bounds__(256) final_kernel(
    const float* __restrict__ g3, const float* __restrict__ r2,
    const float* __restrict__ scsh, float* __restrict__ out)
{
    size_t idx = (size_t)blockIdx.x * 256 + threadIdx.x;
    int w = (int)(idx % W_);
    size_t t = idx / W_;
    int h = (int)(t % H_); t /= H_;
    int c = (int)(t % C_);
    int b = (int)(t / C_);
    size_t m = (size_t)(b * NTOK + h * W_ + w) * C_ + c;
    float v = g3[m] * scsh[c] + scsh[C_ + c];
    out[idx] = r2[m] + gelu_f(v);
}

// ---------------------------------------------------------------------------
// Host orchestration
// ---------------------------------------------------------------------------
extern "C" void kernel_launch(void* const* d_in, const int* in_sizes, int n_in,
                              void* d_out, int out_size, void* d_ws, size_t ws_size,
                              hipStream_t stream) {
    const float* x      = (const float*)d_in[0];
    const float* ln1_w  = (const float*)d_in[1];
    const float* ln1_b  = (const float*)d_in[2];
    const float* pos_w1 = (const float*)d_in[3];
    const float* pos_w2 = (const float*)d_in[4];
    const float* qkv_w  = (const float*)d_in[5];
    const float* qkv_b  = (const float*)d_in[6];
    const float* rpb    = (const float*)d_in[7];
    const float* proj_w = (const float*)d_in[8];
    const float* proj_b = (const float*)d_in[9];
    const float* ln2_w  = (const float*)d_in[10];
    const float* ln2_b  = (const float*)d_in[11];
    const float* fc1_w  = (const float*)d_in[12];
    const float* fc1_b  = (const float*)d_in[13];
    const float* bn1_g  = (const float*)d_in[14];
    const float* bn1_b  = (const float*)d_in[15];
    const float* dw_w   = (const float*)d_in[16];
    const float* dw_b   = (const float*)d_in[17];
    const float* bn2_g  = (const float*)d_in[18];
    const float* bn2_b  = (const float*)d_in[19];
    const float* fc2_w  = (const float*)d_in[20];
    const float* fc2_b  = (const float*)d_in[21];
    const float* bn3_g  = (const float*)d_in[22];
    const float* bn3_b  = (const float*)d_in[23];
    float* out = (float*)d_out;

    char* ws = (char*)d_ws;
    const size_t SZ_R1 = (size_t)MROWS * C_ * 4;      // 51,380,224
    const size_t SZ_TB = (size_t)MROWS * C_ * 2;      // 25,690,112
    float* r1   = (float*)(ws);                       // residual 1 / g3
    float* r2   = (float*)(ws + SZ_R1);               // residual 2
    bf16*  tbuf = (bf16*)(ws + 2 * SZ_R1);            // t / attn-out / ln2-out
    bf16*  wb   = (bf16*)(ws + 2 * SZ_R1 + SZ_TB);    // bf16 weights (786432 elems)
    float* st   = (float*)(ws + 2 * SZ_R1 + SZ_TB + 786432 * 2);  // stats (64 KB)
    char*  big  = ws + 2 * SZ_R1 + SZ_TB + 786432 * 2 + 65536;
    bf16*  qkvh = (bf16*)big;                         // M x 768
    bf16*  g1   = (bf16*)big;                         // M x 1024 (after attn)
    bf16*  h1   = (bf16*)(big + (size_t)MROWS * HID_ * 2);  // M x 1024

    bf16* qkvW  = wb;
    bf16* projW = wb + 196608;
    bf16* fc1W  = wb + 262144;
    bf16* fc2W  = wb + 524288;
    float* st1 = st, *st2 = st + 2048, *st3 = st + 4096;
    float* sc1 = st + 6144, *sc2 = st + 8192, *sc3 = st + 10240;

    const float inv_n = 1.0f / (float)MROWS;

    // weights -> bf16
    cvt_bf16_kernel<<<768, 256, 0, stream>>>(qkv_w, qkvW, 196608);
    cvt_bf16_kernel<<<256, 256, 0, stream>>>(proj_w, projW, 65536);
    cvt_bf16_kernel<<<1024, 256, 0, stream>>>(fc1_w, fc1W, 262144);
    cvt_bf16_kernel<<<1024, 256, 0, stream>>>(fc2_w, fc2W, 262144);

    // LN1 + window partition + LePE convs
    ln1_pos_kernel<<<NWIN, 256, 0, stream>>>(x, ln1_w, ln1_b, pos_w1, pos_w2, r1, tbuf);

    // QKV GEMM: (50176 x 256) x (256 x 768), bf16 out (window-ordered rows)
    gemm_wmma_bf16<<<dim3(MROWS / 64, 768 / 128), dim3(32, 4), 0, stream>>>(
        tbuf, qkvW, qkv_b, nullptr, nullptr, qkvh, MROWS, 3 * C_, C_);

    // attention per (window, head); output de-windowed into tbuf
    attn_kernel<<<dim3(NWIN, NH_), 64, 0, stream>>>(qkvh, rpb, tbuf);

    // proj GEMM + fused residual: r2 = r1 + o @ proj_w^T + proj_b
    gemm_wmma_bf16<<<dim3(MROWS / 64, C_ / 128), dim3(32, 4), 0, stream>>>(
        tbuf, projW, proj_b, r1, r2, nullptr, MROWS, C_, C_);

    // LN2 -> tbuf (bf16)
    ln2_kernel<<<MROWS / 8, 256, 0, stream>>>(r2, ln2_w, ln2_b, tbuf);

    // fc1 GEMM: (50176 x 256) x (256 x 1024) -> g1 (bf16)
    gemm_wmma_bf16<<<dim3(MROWS / 64, HID_ / 128), dim3(32, 4), 0, stream>>>(
        tbuf, fc1W, fc1_b, nullptr, nullptr, g1, MROWS, HID_, C_);

    // zero BN stats (re-zeroed every launch; graph-replay safe)
    zero_f32_kernel<<<48, 256, 0, stream>>>(st, 12288);

    // BN1 + GELU
    bn_stats_bf16_kernel<<<MROWS / 56, 256, 0, stream>>>(g1, st1, HID_, 56);
    bn_finalize_kernel<<<HID_ / 256, 256, 0, stream>>>(st1, bn1_g, bn1_b, sc1, HID_, inv_n);
    bn_gelu_bf16_kernel<<<(unsigned)((size_t)MROWS * HID_ / 256), 256, 0, stream>>>(g1, sc1, h1, HID_);

    // depthwise 3x3 + bias -> back into g1 buffer
    dwconv_kernel<<<MROWS, 256, 0, stream>>>(h1, dw_w, dw_b, g1);

    // BN2 + GELU -> h1 buffer
    bn_stats_bf16_kernel<<<MROWS / 56, 256, 0, stream>>>(g1, st2, HID_, 56);
    bn_finalize_kernel<<<HID_ / 256, 256, 0, stream>>>(st2, bn2_g, bn2_b, sc2, HID_, inv_n);
    bn_gelu_bf16_kernel<<<(unsigned)((size_t)MROWS * HID_ / 256), 256, 0, stream>>>(g1, sc2, h1, HID_);

    // fc2 GEMM: (50176 x 1024) x (1024 x 256) -> r1 (f32, reused as g3)
    gemm_wmma_bf16<<<dim3(MROWS / 64, C_ / 128), dim3(32, 4), 0, stream>>>(
        h1, fc2W, fc2_b, nullptr, r1, nullptr, MROWS, C_, HID_);

    // BN3 + GELU + residual + NLC->NCHW
    bn_stats_f32_kernel<<<MROWS / 56, 256, 0, stream>>>(r1, st3, C_, 56);
    bn_finalize_kernel<<<1, 256, 0, stream>>>(st3, bn3_g, bn3_b, sc3, C_, inv_n);
    final_kernel<<<(unsigned)((size_t)B_ * C_ * NTOK / 256), 256, 0, stream>>>(r1, r2, sc3, out);
}
